// MolecularGNNv6_10771777978546
// MI455X (gfx1250) — matrix operand order repaired and checked by reading, not compiled, not run on hardware
//
#include <hip/hip_runtime.h>
#include <hip/hip_bf16.h>
#include <math.h>

#define HIDC 128
#define HEADS 4
#define EDIM 144
#define NGRAPH 2048
#define ENC_NEG_INF 0x007FFFFFu

typedef __attribute__((ext_vector_type(16))) __bf16 v16bf;
typedef __attribute__((ext_vector_type(8)))  float  v8f;

struct bfrag { uint4 lo, hi; };   // 32 bytes = one v16bf fragment

__device__ __forceinline__ unsigned short bf16bits(float f) {
  unsigned u = __builtin_bit_cast(unsigned, f);
  u += 0x7FFFu + ((u >> 16) & 1u);                    // round-to-nearest-even
  return (unsigned short)(u >> 16);
}

// monotone float <-> uint encoding for atomicMax-based segment max
__device__ __forceinline__ unsigned fenc(float f) {
  unsigned u = __builtin_bit_cast(unsigned, f);
  return (u & 0x80000000u) ? ~u : (u | 0x80000000u);
}
__device__ __forceinline__ float fdec(unsigned e) {
  unsigned u = (e & 0x80000000u) ? (e & 0x7FFFFFFFu) : ~e;
  return __builtin_bit_cast(float, u);
}
__device__ __forceinline__ float sigmoidf_(float x) { return 1.0f / (1.0f + __expf(-x)); }

// ---------------------------------------------------------------------------
// Packing kernels: do the f32->bf16 conversion ONCE, off the WMMA hot loop.
// ---------------------------------------------------------------------------
// A: [M,K] f32 row-major -> [M,K] bf16 row-major (2 elems/thread, packed store)
__global__ void k_pack_bf16(const float* __restrict__ src, unsigned* __restrict__ dst, int n2) {
  int t = blockIdx.x * blockDim.x + threadIdx.x;
  if (t >= n2) return;
  float2 v = ((const float2*)src)[t];
  dst[t] = ((unsigned)bf16bits(v.y) << 16) | (unsigned)bf16bits(v.x);
}

// B: [K,N] f32 row-major -> per-lane WMMA fragment layout (ISA 7.12.2):
// Bp[((nt*(K/32)+kt)*32 + lane)*16 + j] = bf16(B[(kt*32 + (lane>>4)*8 + (j&7) + (j>>3)*16), nt*16 + (lane&15)])
__global__ void k_pack_Bfrag(const float* __restrict__ B, unsigned short* __restrict__ Bp,
                             int N, int K) {
  int t = blockIdx.x * blockDim.x + threadIdx.x;
  if (t >= N * K) return;
  int j  = t & 15;
  int l  = (t >> 4) & 31;
  int kt = (t >> 9) % (K >> 5);
  int nt = t / ((K >> 5) << 9);
  int k   = kt * 32 + ((l >> 4) << 3) + (j & 7) + ((j >> 3) << 4);
  int col = nt * 16 + (l & 15);
  Bp[t] = bf16bits(B[(size_t)k * N + col]);
}

// ---------------------------------------------------------------------------
// WMMA bf16 GEMM: C[M,N] (+)= A[M,K] @ B[K,N].
// A pre-converted bf16 row-major; B pre-packed fragments. wave32: one 16x32
// output tile per wave (2 accumulators reuse the A fragment). Inner loop is
// pure b128 loads + v_wmma_f32_16x16x32_bf16.
// ---------------------------------------------------------------------------
__global__ void k_gemm_bf16(const unsigned short* __restrict__ A,
                            const unsigned short* __restrict__ Bp,
                            float* __restrict__ C, int ldc,
                            int M, int N, int K, int accumulate) {
  const int lane = threadIdx.x;          // 0..31
  const int wave = threadIdx.y;          // 0..7
  const int n0 = blockIdx.x * 32;
  const int m0 = (blockIdx.y * 8 + wave) * 16;
  if (m0 >= M) return;
  const int half = lane >> 4;
  const int l16  = lane & 15;
  const int row  = m0 + l16;
  const int K32  = K >> 5;

  v8f acc0, acc1;
  if (accumulate) {
#pragma unroll
    for (int v = 0; v < 8; ++v) {
      acc0[v] = C[(size_t)(m0 + v + 8 * half) * ldc + n0 + l16];
      acc1[v] = C[(size_t)(m0 + v + 8 * half) * ldc + n0 + 16 + l16];
    }
  } else {
#pragma unroll
    for (int v = 0; v < 8; ++v) { acc0[v] = 0.0f; acc1[v] = 0.0f; }
  }

  const unsigned short* arow = A + (size_t)row * K;
  const unsigned short* bp0  = Bp + ((size_t)(n0 >> 4) * K32) * 512 + lane * 16;
  const unsigned short* bp1  = bp0 + (size_t)K32 * 512;

  for (int kt = 0; kt < K32; ++kt) {
    // A fragment: lane holds K = kt*32 + half*8 + {0..7} and +16 + {0..7}
    const unsigned short* ap = arow + kt * 32 + half * 8;
    __builtin_prefetch(ap + 32, 0, 1);                 // global_prefetch_b8
    bfrag fa;
    fa.lo = *(const uint4*)(ap);
    fa.hi = *(const uint4*)(ap + 16);
    v16bf af = __builtin_bit_cast(v16bf, fa);

    bfrag fb0, fb1;
    const unsigned short* b0 = bp0 + (size_t)kt * 512;
    const unsigned short* b1 = bp1 + (size_t)kt * 512;
    fb0.lo = *(const uint4*)(b0); fb0.hi = *(const uint4*)(b0 + 8);
    fb1.lo = *(const uint4*)(b1); fb1.hi = *(const uint4*)(b1 + 8);
    v16bf bf0 = __builtin_bit_cast(v16bf, fb0);
    v16bf bf1 = __builtin_bit_cast(v16bf, fb1);

    acc0 = __builtin_amdgcn_wmma_f32_16x16x32_bf16(false, af, false, bf0,
                                                   (short)0, acc0, false, false);
    acc1 = __builtin_amdgcn_wmma_f32_16x16x32_bf16(false, af, false, bf1,
                                                   (short)0, acc1, false, false);
  }
#pragma unroll
  for (int v = 0; v < 8; ++v) {
    C[(size_t)(m0 + v + 8 * half) * ldc + n0 + l16]      = acc0[v];
    C[(size_t)(m0 + v + 8 * half) * ldc + n0 + 16 + l16] = acc1[v];
  }
}

// ---------------------------------------------------------------------------
__global__ void k_fill_u32(unsigned* p, unsigned v, int n) {
  int t = blockIdx.x * blockDim.x + threadIdx.x;
  if (t < n) p[t] = v;
}

// h0 = x @ atom_W + atom_b   (block = one node row, K=47 staged in LDS)
__global__ void k_node_embed(const float* __restrict__ x, const float* __restrict__ W,
                             const float* __restrict__ b, float* __restrict__ h0, int N) {
  __shared__ float xs[47];
  int n = blockIdx.x, c = threadIdx.x;
  if (c < 47) xs[c] = x[(size_t)n * 47 + c];
  __syncthreads();
  float s = b[c];
  for (int k = 0; k < 47; ++k) s += xs[k] * W[k * HIDC + c];
  h0[(size_t)n * HIDC + c] = s;
}

// ef = [edge_attr @ bond_W + bond_b , RBF(d)]  (block = one edge, 144 threads)
__global__ void k_edge_feat(const float* __restrict__ ea, const float* __restrict__ bW,
                            const float* __restrict__ bb, float* __restrict__ ef, int E) {
  __shared__ float es[13];
  int e = blockIdx.x, c = threadIdx.x;
  if (c < 13) es[c] = ea[(size_t)e * 13 + c];
  __syncthreads();
  float v;
  if (c < HIDC) {
    v = bb[c];
    for (int k = 0; k < 13; ++k) v += es[k] * bW[k * HIDC + c];
  } else {
    int i = c - HIDC;
    float d = es[12];
    float ctr = 10.0f * (float)i / 15.0f;
    float diff = d - ctr;
    v = __expf(-diff * diff / (2.0f * 0.3125f * 0.3125f));
  }
  ef[(size_t)e * EDIM + c] = v;
}

// column mean of ef over edges (grid = 144 channels)
__global__ void k_colmean(const float* __restrict__ ef, float* __restrict__ mean, int E) {
  int c = blockIdx.x;
  float s = 0.0f;
  for (int e = threadIdx.x; e < E; e += blockDim.x) s += ef[(size_t)e * EDIM + c];
  __shared__ float red[256];
  red[threadIdx.x] = s; __syncthreads();
  for (int o = 128; o > 0; o >>= 1) {
    if ((int)threadIdx.x < o) red[threadIdx.x] += red[threadIdx.x + o];
    __syncthreads();
  }
  if (threadIdx.x == 0) mean[c] = red[0] / (float)E;
}

// wE[c,h] = sum_j edge_W[c, h*128+j] * att_edge[h,j]   (contract edge GEMM away)
__global__ void k_edge_w(const float* __restrict__ eW, const float* __restrict__ attE,
                         float* __restrict__ wE) {
  int t = blockIdx.x * blockDim.x + threadIdx.x;
  if (t >= EDIM * HEADS) return;
  int c = t / HEADS, h = t % HEADS;
  float s = 0.0f;
  for (int j = 0; j < HIDC; ++j)
    s += eW[(size_t)c * (HEADS * HIDC) + h * HIDC + j] * attE[h * HIDC + j];
  wE[c * HEADS + h] = s;
}

__global__ void k_loop_ae(const float* __restrict__ mean, const float* __restrict__ wE,
                          float* __restrict__ aeLoop) {
  int h = threadIdx.x;
  if (h >= HEADS) return;
  float s = 0.0f;
  for (int c = 0; c < EDIM; ++c) s += mean[c] * wE[c * HEADS + h];
  aeLoop[h] = s;
}

// a_s/a_d from materialized xt
__global__ void k_node_attn(const float* __restrict__ xt, const float* __restrict__ attS,
                            const float* __restrict__ attD, float* __restrict__ as_,
                            float* __restrict__ ad_, int N) {
  int t = blockIdx.x * blockDim.x + threadIdx.x;
  if (t >= N * HEADS) return;
  int n = t / HEADS, h = t % HEADS;
  const float* xp = xt + (size_t)n * (HEADS * HIDC) + h * HIDC;
  float s = 0.0f, d = 0.0f;
  for (int j = 0; j < HIDC; ++j) { float v = xp[j]; s += v * attS[h * HIDC + j]; d += v * attD[h * HIDC + j]; }
  as_[t] = s; ad_[t] = d;
}

__global__ void k_edge_ae(const float* __restrict__ ef, const float* __restrict__ wE,
                          float* __restrict__ ae, int E) {
  int t = blockIdx.x * blockDim.x + threadIdx.x;
  if (t >= E * HEADS) return;
  int e = t / HEADS, h = t % HEADS;
  const float* ep = ef + (size_t)e * EDIM;
  float s = 0.0f;
  for (int c = 0; c < EDIM; ++c) s += ep[c] * wE[c * HEADS + h];
  ae[t] = s;
}

__global__ void k_logits_max(const int* __restrict__ ei, const float* __restrict__ as_,
                             const float* __restrict__ ad_, const float* __restrict__ ae,
                             const float* __restrict__ aeLoop, float* __restrict__ raw,
                             unsigned* __restrict__ segmax, int E, int N) {
  int t = blockIdx.x * blockDim.x + threadIdx.x;
  if (t >= (E + N) * HEADS) return;
  int e = t / HEADS, h = t % HEADS;
  int s, d; float a;
  if (e < E) { s = ei[e]; d = ei[E + e]; a = ae[t]; }
  else       { s = d = e - E; a = aeLoop[h]; }
  float v = as_[s * HEADS + h] + ad_[d * HEADS + h] + a;
  v = v > 0.0f ? v : 0.2f * v;                         // leaky_relu(0.2)
  raw[t] = v;
  atomicMax(&segmax[d * HEADS + h], fenc(v));
}

__global__ void k_exp_sum(const int* __restrict__ ei, float* __restrict__ raw,
                          const unsigned* __restrict__ segmax, float* __restrict__ segden,
                          int E, int N) {
  int t = blockIdx.x * blockDim.x + threadIdx.x;
  if (t >= (E + N) * HEADS) return;
  int e = t / HEADS, h = t % HEADS;
  int d = (e < E) ? ei[E + e] : (e - E);
  float ex = __expf(raw[t] - fdec(segmax[d * HEADS + h]));
  raw[t] = ex;
  atomicAdd(&segden[d * HEADS + h], ex);
}

// wave per (edge, head): scatter-add alpha/4 * xt[src,h,:] into agg[dst,:]
__global__ void k_aggregate(const int* __restrict__ ei, const float* __restrict__ raw,
                            const float* __restrict__ segden, const float* __restrict__ xt,
                            float* __restrict__ agg, int E, int N) {
  int gt = blockIdx.x * blockDim.x + threadIdx.x;
  int w = gt >> 5, lane = gt & 31;
  if (w >= (E + N) * HEADS) return;
  int e = w / HEADS, h = w % HEADS;
  int s, d;
  if (e < E) { s = ei[e]; d = ei[E + e]; } else { s = d = e - E; }
  float wt = raw[w] / segden[d * HEADS + h] * 0.25f;   // mean over heads folded in
  const float* xp = xt + (size_t)s * (HEADS * HIDC) + h * HIDC;
  float* op = agg + (size_t)d * HIDC;
#pragma unroll
  for (int q = 0; q < 4; ++q) {
    int c = lane + q * 32;
    atomicAdd(&op[c], wt * xp[c]);
  }
}

__global__ void k_bn_stats(const float* __restrict__ agg, const float* __restrict__ bias,
                           float* __restrict__ mean, float* __restrict__ var, int N) {
  int c = blockIdx.x;
  float b = bias[c], s = 0.0f, s2 = 0.0f;
  for (int n = threadIdx.x; n < N; n += blockDim.x) {
    float v = agg[(size_t)n * HIDC + c] + b;
    s += v; s2 += v * v;
  }
  __shared__ float r1[256], r2[256];
  r1[threadIdx.x] = s; r2[threadIdx.x] = s2; __syncthreads();
  for (int o = 128; o > 0; o >>= 1) {
    if ((int)threadIdx.x < o) { r1[threadIdx.x] += r1[threadIdx.x + o]; r2[threadIdx.x] += r2[threadIdx.x + o]; }
    __syncthreads();
  }
  if (threadIdx.x == 0) {
    float mu = r1[0] / (float)N;
    mean[c] = mu;
    var[c] = r2[0] / (float)N - mu * mu;
  }
}

__global__ void k_bn_apply(const float* __restrict__ agg, const float* __restrict__ bias,
                           const float* __restrict__ mean, const float* __restrict__ var,
                           const float* __restrict__ gamma, const float* __restrict__ beta,
                           const float* __restrict__ hprev, float* __restrict__ hnext, int N) {
  int t = blockIdx.x * blockDim.x + threadIdx.x;
  if (t >= N * HIDC) return;
  int c = t % HIDC;
  float v = agg[t] + bias[c];
  float y = gamma[c] * (v - mean[c]) * rsqrtf(var[c] + 1e-5f) + beta[c];
  y = y > 0.0f ? y : (__expf(y) - 1.0f);               // ELU
  hnext[t] = y + hprev[t];
}

__global__ void k_layernorm(float* __restrict__ jk, const float* __restrict__ jb,
                            const float* __restrict__ g, const float* __restrict__ b, int N) {
  int n = blockIdx.x, c = threadIdx.x;
  float v = jk[(size_t)n * HIDC + c] + jb[c];
  __shared__ float r1[128], r2[128];
  r1[c] = v; r2[c] = v * v; __syncthreads();
  for (int o = 64; o > 0; o >>= 1) {
    if (c < o) { r1[c] += r1[c + o]; r2[c] += r2[c + o]; }
    __syncthreads();
  }
  float mu = r1[0] / (float)HIDC;
  float var = r2[0] / (float)HIDC - mu * mu;
  jk[(size_t)n * HIDC + c] = g[c] * (v - mu) * rsqrtf(var + 1e-5f) + b[c];
}

__global__ void k_lstm_gates(const float* __restrict__ qstar, const float* __restrict__ hh,
                             const float* __restrict__ Wih, const float* __restrict__ Whh,
                             const float* __restrict__ bih, const float* __restrict__ bhh,
                             float* __restrict__ gates) {
  int t = blockIdx.x * blockDim.x + threadIdx.x;
  if (t >= NGRAPH * 4 * HIDC) return;
  int bg = t / (4 * HIDC), r = t % (4 * HIDC);
  const float* q  = qstar + (size_t)bg * 2 * HIDC;
  const float* hp = hh + (size_t)bg * HIDC;
  const float* wi = Wih + (size_t)r * 2 * HIDC;
  const float* wh = Whh + (size_t)r * HIDC;
  float s = bih[r] + bhh[r];
  for (int j = 0; j < 2 * HIDC; ++j) s += q[j] * wi[j];
  for (int j = 0; j < HIDC; ++j)     s += hp[j] * wh[j];
  gates[t] = s;
}

__global__ void k_lstm_update(const float* __restrict__ gates, float* __restrict__ cc,
                              float* __restrict__ hh) {
  int t = blockIdx.x * blockDim.x + threadIdx.x;
  if (t >= NGRAPH * HIDC) return;
  int bg = t / HIDC, j = t % HIDC;
  const float* gp = gates + (size_t)bg * 4 * HIDC;
  float i = gp[j], f = gp[HIDC + j], gc = gp[2 * HIDC + j], o = gp[3 * HIDC + j];
  float c = sigmoidf_(f) * cc[t] + sigmoidf_(i) * tanhf(gc);
  cc[t] = c;
  hh[t] = sigmoidf_(o) * tanhf(c);
}

// wave per node: score = <jk[n], hh[batch[n]]>, segment max via atomicMax
__global__ void k_scores_max(const float* __restrict__ jk, const float* __restrict__ hh,
                             const int* __restrict__ batch, float* __restrict__ scores,
                             unsigned* __restrict__ smax, int N) {
  int gt = blockIdx.x * blockDim.x + threadIdx.x;
  int w = gt >> 5, lane = gt & 31;
  if (w >= N) return;
  int b = batch[w];
  const float* jp = jk + (size_t)w * HIDC;
  const float* hp = hh + (size_t)b * HIDC;
  float s = 0.0f;
#pragma unroll
  for (int q = 0; q < 4; ++q) { int c = lane + q * 32; s += jp[c] * hp[c]; }
  for (int o = 16; o > 0; o >>= 1) s += __shfl_down(s, o, 32);
  if (lane == 0) { scores[w] = s; atomicMax(&smax[b], fenc(s)); }
}

__global__ void k_scores_exp(float* __restrict__ scores, const int* __restrict__ batch,
                             const unsigned* __restrict__ smax, float* __restrict__ sden, int N) {
  int n = blockIdx.x * blockDim.x + threadIdx.x;
  if (n >= N) return;
  int b = batch[n];
  float ex = __expf(scores[n] - fdec(smax[b]));
  scores[n] = ex;
  atomicAdd(&sden[b], ex);
}

__global__ void k_s2s_agg(const float* __restrict__ scores, const float* __restrict__ sden,
                          const int* __restrict__ batch, const float* __restrict__ jk,
                          float* __restrict__ r, int N) {
  int gt = blockIdx.x * blockDim.x + threadIdx.x;
  int w = gt >> 5, lane = gt & 31;
  if (w >= N) return;
  int b = batch[w];
  float a = scores[w] / sden[b];
  const float* jp = jk + (size_t)w * HIDC;
  float* rp = r + (size_t)b * HIDC;
#pragma unroll
  for (int q = 0; q < 4; ++q) { int c = lane + q * 32; atomicAdd(&rp[c], a * jp[c]); }
}

__global__ void k_qstar(const float* __restrict__ hh, const float* __restrict__ r,
                        float* __restrict__ qstar) {
  int t = blockIdx.x * blockDim.x + threadIdx.x;
  if (t >= NGRAPH * 2 * HIDC) return;
  int b = t / (2 * HIDC), j = t % (2 * HIDC);
  qstar[t] = (j < HIDC) ? hh[(size_t)b * HIDC + j] : r[(size_t)b * HIDC + (j - HIDC)];
}

__global__ void k_readout(const float* __restrict__ qstar, const float* __restrict__ W,
                          const float* __restrict__ bias, float* __restrict__ out) {
  int t = blockIdx.x * blockDim.x + threadIdx.x;
  if (t >= NGRAPH * HIDC) return;
  int b = t / HIDC, c = t % HIDC;
  const float* q = qstar + (size_t)b * 2 * HIDC;
  float s = bias[c];
  for (int j = 0; j < 2 * HIDC; ++j) s += q[j] * W[j * HIDC + c];
  out[t] = s;
}

// ---------------------------------------------------------------------------
extern "C" void kernel_launch(void* const* d_in, const int* in_sizes, int n_in,
                              void* d_out, int out_size, void* d_ws, size_t ws_size,
                              hipStream_t stream) {
  (void)n_in; (void)out_size; (void)ws_size;
  const float* x            = (const float*)d_in[0];
  const float* edge_attr    = (const float*)d_in[1];
  const int*   edge_index   = (const int*)  d_in[2];
  const int*   batch        = (const int*)  d_in[3];
  const float* atom_W       = (const float*)d_in[4];
  const float* atom_b       = (const float*)d_in[5];
  const float* bond_W       = (const float*)d_in[6];
  const float* bond_b       = (const float*)d_in[7];
  const float* conv_lin_W   = (const float*)d_in[8];
  const float* conv_att_src = (const float*)d_in[9];
  const float* conv_att_dst = (const float*)d_in[10];
  const float* conv_att_edge= (const float*)d_in[11];
  const float* conv_edge_W  = (const float*)d_in[12];
  const float* conv_bias    = (const float*)d_in[13];
  const float* bn_gamma     = (const float*)d_in[14];
  const float* bn_beta      = (const float*)d_in[15];
  const float* jk_W         = (const float*)d_in[16];
  const float* jk_b         = (const float*)d_in[17];
  const float* ln_gamma     = (const float*)d_in[18];
  const float* ln_beta      = (const float*)d_in[19];
  const float* lstm_Wih     = (const float*)d_in[20];
  const float* lstm_Whh     = (const float*)d_in[21];
  const float* lstm_bih     = (const float*)d_in[22];
  const float* lstm_bhh     = (const float*)d_in[23];
  const float* ro_W         = (const float*)d_in[24];
  const float* ro_b         = (const float*)d_in[25];

  const int N = in_sizes[0] / 47;    // 60000
  const int E = in_sizes[1] / 13;    // 150000

  // ---- workspace carve (bump allocator, 256B aligned) ----
  char* p = (char*)d_ws;
  auto alloc = [&](size_t bytes) -> void* {
    void* r = (void*)p;
    p += (bytes + 255) & ~(size_t)255;
    return r;
  };
  float*    ef       = (float*)   alloc((size_t)E * EDIM * 4);
  float*    loopmean = (float*)   alloc(EDIM * 4);
  float*    hbuf[4];
  for (int i = 0; i < 4; ++i) hbuf[i] = (float*)alloc((size_t)N * HIDC * 4);
  float*    xt       = (float*)   alloc((size_t)N * HEADS * HIDC * 4);
  float*    agg      = (float*)   alloc((size_t)N * HIDC * 4);
  float*    as_      = (float*)   alloc((size_t)N * HEADS * 4);
  float*    ad_      = (float*)   alloc((size_t)N * HEADS * 4);
  float*    ae       = (float*)   alloc((size_t)E * HEADS * 4);
  float*    aeLoop   = (float*)   alloc(HEADS * 4);
  float*    wE       = (float*)   alloc(EDIM * HEADS * 4);
  float*    raw      = (float*)   alloc((size_t)(E + N) * HEADS * 4);
  unsigned* segmax   = (unsigned*)alloc((size_t)N * HEADS * 4);
  float*    segden   = (float*)   alloc((size_t)N * HEADS * 4);
  float*    bn_mu    = (float*)   alloc(HIDC * 4);
  float*    bn_var   = (float*)   alloc(HIDC * 4);
  float*    jk       = (float*)   alloc((size_t)N * HIDC * 4);
  float*    qstar    = (float*)   alloc((size_t)NGRAPH * 2 * HIDC * 4);
  float*    hhb      = (float*)   alloc((size_t)NGRAPH * HIDC * 4);
  float*    ccb      = (float*)   alloc((size_t)NGRAPH * HIDC * 4);
  float*    gates    = (float*)   alloc((size_t)NGRAPH * 4 * HIDC * 4);
  float*    scores   = (float*)   alloc((size_t)N * 4);
  unsigned* smax2    = (unsigned*)alloc(NGRAPH * 4);
  float*    sden2    = (float*)   alloc(NGRAPH * 4);
  float*    rbuf     = (float*)   alloc((size_t)NGRAPH * HIDC * 4);
  unsigned short* packA = (unsigned short*)alloc((size_t)N * HIDC * 2);           // bf16 A
  unsigned short* packB = (unsigned short*)alloc((size_t)HIDC * HEADS * HIDC * 2);// bf16 B frags

  const int TOT = (E + N) * HEADS;
  const dim3 wgb(32, 8);                               // 8 waves / block

  // ---- encoders ----
  k_node_embed<<<N, 128, 0, stream>>>(x, atom_W, atom_b, hbuf[0], N);
  k_edge_feat<<<E, 144, 0, stream>>>(edge_attr, bond_W, bond_b, ef, E);
  k_colmean<<<144, 256, 0, stream>>>(ef, loopmean, E);

  // ---- 3 GAT layers ----
  for (int l = 0; l < 3; ++l) {
    const float* linW = conv_lin_W + (size_t)l * HIDC * (HEADS * HIDC);
    k_edge_w<<<(EDIM * HEADS + 127) / 128, 128, 0, stream>>>(
        conv_edge_W + (size_t)l * EDIM * (HEADS * HIDC),
        conv_att_edge + (size_t)l * HEADS * HIDC, wE);
    k_loop_ae<<<1, HEADS, 0, stream>>>(loopmean, wE, aeLoop);

    // xt = h_l @ lin_W  via WMMA (pack once, matrix-core loop)
    k_pack_bf16<<<(N * HIDC / 2 + 255) / 256, 256, 0, stream>>>(hbuf[l], (unsigned*)packA,
                                                                N * HIDC / 2);
    k_pack_Bfrag<<<(HIDC * HEADS * HIDC + 255) / 256, 256, 0, stream>>>(linW, packB,
                                                                        HEADS * HIDC, HIDC);
    dim3 gg((HEADS * HIDC) / 32, (N / 16 + 7) / 8);
    k_gemm_bf16<<<gg, wgb, 0, stream>>>(packA, packB, xt, HEADS * HIDC,
                                        N, HEADS * HIDC, HIDC, 0);

    k_node_attn<<<(N * HEADS + 255) / 256, 256, 0, stream>>>(
        xt, conv_att_src + (size_t)l * HEADS * HIDC,
        conv_att_dst + (size_t)l * HEADS * HIDC, as_, ad_, N);
    k_edge_ae<<<(E * HEADS + 255) / 256, 256, 0, stream>>>(ef, wE, ae, E);

    k_fill_u32<<<(N * HEADS + 255) / 256, 256, 0, stream>>>(segmax, ENC_NEG_INF, N * HEADS);
    hipMemsetAsync(segden, 0, (size_t)N * HEADS * 4, stream);
    hipMemsetAsync(agg, 0, (size_t)N * HIDC * 4, stream);

    k_logits_max<<<(TOT + 255) / 256, 256, 0, stream>>>(edge_index, as_, ad_, ae, aeLoop,
                                                        raw, segmax, E, N);
    k_exp_sum<<<(TOT + 255) / 256, 256, 0, stream>>>(edge_index, raw, segmax, segden, E, N);
    long lanes = (long)TOT * 32;
    k_aggregate<<<(unsigned)((lanes + 255) / 256), 256, 0, stream>>>(edge_index, raw, segden,
                                                                     xt, agg, E, N);
    k_bn_stats<<<HIDC, 256, 0, stream>>>(agg, conv_bias + l * HIDC, bn_mu, bn_var, N);
    k_bn_apply<<<(N * HIDC + 255) / 256, 256, 0, stream>>>(
        agg, conv_bias + l * HIDC, bn_mu, bn_var, bn_gamma + l * HIDC, bn_beta + l * HIDC,
        hbuf[l], hbuf[l + 1], N);
  }

  // ---- JK projection (split 384-K into three accumulating WMMA GEMMs) + LayerNorm ----
  {
    dim3 gj(HIDC / 32, (N / 16 + 7) / 8);
    for (int l = 0; l < 3; ++l) {
      k_pack_bf16<<<(N * HIDC / 2 + 255) / 256, 256, 0, stream>>>(hbuf[l + 1], (unsigned*)packA,
                                                                  N * HIDC / 2);
      k_pack_Bfrag<<<(HIDC * HIDC + 255) / 256, 256, 0, stream>>>(
          jk_W + (size_t)l * HIDC * HIDC, packB, HIDC, HIDC);
      k_gemm_bf16<<<gj, wgb, 0, stream>>>(packA, packB, jk, HIDC, N, HIDC, HIDC, l > 0);
    }
    k_layernorm<<<N, 128, 0, stream>>>(jk, jk_b, ln_gamma, ln_beta, N);
  }

  // ---- Set2Set (3 iterations) ----
  hipMemsetAsync(qstar, 0, (size_t)NGRAPH * 2 * HIDC * 4, stream);
  hipMemsetAsync(hhb, 0, (size_t)NGRAPH * HIDC * 4, stream);
  hipMemsetAsync(ccb, 0, (size_t)NGRAPH * HIDC * 4, stream);
  for (int it = 0; it < 3; ++it) {
    k_lstm_gates<<<(NGRAPH * 4 * HIDC + 255) / 256, 256, 0, stream>>>(
        qstar, hhb, lstm_Wih, lstm_Whh, lstm_bih, lstm_bhh, gates);
    k_lstm_update<<<(NGRAPH * HIDC + 255) / 256, 256, 0, stream>>>(gates, ccb, hhb);
    k_fill_u32<<<(NGRAPH + 255) / 256, 256, 0, stream>>>(smax2, ENC_NEG_INF, NGRAPH);
    hipMemsetAsync(sden2, 0, NGRAPH * 4, stream);
    hipMemsetAsync(rbuf, 0, (size_t)NGRAPH * HIDC * 4, stream);
    k_scores_max<<<((long)N * 32 + 255) / 256, 256, 0, stream>>>(jk, hhb, batch, scores, smax2, N);
    k_scores_exp<<<(N + 255) / 256, 256, 0, stream>>>(scores, batch, smax2, sden2, N);
    k_s2s_agg<<<((long)N * 32 + 255) / 256, 256, 0, stream>>>(scores, sden2, batch, jk, rbuf, N);
    k_qstar<<<(NGRAPH * 2 * HIDC + 255) / 256, 256, 0, stream>>>(hhb, rbuf, qstar);
  }

  // ---- readout ----
  k_readout<<<(NGRAPH * HIDC + 255) / 256, 256, 0, stream>>>(qstar, ro_W, ro_b, (float*)d_out);
}
